// ScaledDotProductAttention_68702296867206
// MI455X (gfx1250) — matrix operand order repaired and checked by reading, not compiled, not run on hardware
//
#include <hip/hip_runtime.h>
#include <stdint.h>

// CDNA5 / gfx1250 flash attention with per-batch forcing reweight.
// S^T = K Q^T and O^T = V^T P^T (both GEMMs transposed) -> per-lane softmax
// state. Scores computed in log2 domain (hardware v_exp_f32 is exp2).
// Scalar (SGPR) branches via readfirstlane; cross-half exchanges via
// v_permlanex16 (VALU) instead of ds_bpermute.

typedef __bf16   v16bf __attribute__((ext_vector_type(16)));
typedef float    v8f   __attribute__((ext_vector_type(8)));
typedef uint32_t u32x4 __attribute__((ext_vector_type(4)));
typedef float    f32x4 __attribute__((ext_vector_type(4)));

#define NEG_BIG (-1.0e9f)

union V16 {
    v16bf    v;
    u32x4    q2[2];
    uint32_t u[8];
    uint16_t e[16];
};

static __device__ __forceinline__ uint16_t bfbits_rne(float f) {
    union { float f; uint32_t u; } c; c.f = f;
    uint32_t r = c.u + 0x7FFFu + ((c.u >> 16) & 1u);
    return (uint16_t)(r >> 16);
}
// Two f32 -> packed bf16 pair (truncate) in ONE v_perm_b32.
static __device__ __forceinline__ uint32_t pack2hi(float a, float b) {
    return __builtin_amdgcn_perm(__float_as_uint(b), __float_as_uint(a), 0x07060302u);
}
static __device__ __forceinline__ float ex2(float x) {
    return __builtin_amdgcn_exp2f(x);                 // bare v_exp_f32
}
// xor-16 cross-half exchange: v_permlanex16_b32 with identity lane select.
static __device__ __forceinline__ uint32_t xor16u(uint32_t x) {
#if __has_builtin(__builtin_amdgcn_permlanex16)
    return (uint32_t)__builtin_amdgcn_permlanex16((int)x, (int)x,
                                                  0x76543210, 0xfedcba98,
                                                  false, false);
#else
    return (uint32_t)__builtin_amdgcn_ds_swizzle((int)x, 0x401f); // SWAPX16
#endif
}
static __device__ __forceinline__ float xor16f(float x) {
    return __uint_as_float(xor16u(__float_as_uint(x)));
}

// One 16(qrows) x 32(keys) attention tile for one wave. Scores in log2 domain.
template<bool MASKED>
static __device__ __forceinline__ void attn_tile(
    const uint16_t (*bK)[64], const uint16_t (*bV)[32],
    const V16* qt, v8f* o, float& m_run, float& l_run,
    int kt0, int i, int lh, int ln, int idxb, float wgt, float l2w)
{
    // ---- S^T = K * Q^T : A = K rows (keys), B = Q^T. 4 WMMA. ----
    v8f acc[2];
    #pragma unroll
    for (int mm = 0; mm < 2; ++mm) {
        v8f c = {};
        #pragma unroll
        for (int t = 0; t < 2; ++t) {
            V16 ka;                                    // A-layout row = key mm*16+ln
            const uint16_t* src = &bK[mm * 16 + ln][t * 32 + lh * 16];
            ka.q2[0] = *(const u32x4*)src;
            ka.q2[1] = *(const u32x4*)(src + 8);
            c = __builtin_amdgcn_wmma_f32_16x16x32_bf16(
                    false, ka.v, false, qt[t].v, (short)0, c, false, false);
        }
        acc[mm] = c;
    }

    // ---- per-lane softmax along key (register) dimension ----
    const int jb = kt0 + 8 * lh;          // key of element (mm,vv): jb + mm*16 + vv
    float s[2][8];
    float mx = NEG_BIG;
    #pragma unroll
    for (int mm = 0; mm < 2; ++mm)
        #pragma unroll
        for (int vv = 0; vv < 8; ++vv) {
            float sv = acc[mm][vv];
            if (MASKED) sv = ((jb + mm * 16 + vv) <= i) ? sv : NEG_BIG;
            s[mm][vv] = sv;
            mx = fmaxf(mx, sv);
        }
    mx = fmaxf(mx, xor16f(mx));
    const float mnew = fmaxf(m_run, mx);
    const float corr = ex2(m_run - mnew);
    m_run = mnew;

    float rs = 0.0f;
    uint32_t pq[2][4];
    const bool allw     = (idxb <= kt0);              // scalar: whole tile weighted
    const bool straddle = !allw && (idxb < kt0 + 32); // scalar: boundary in tile
    if (straddle) {                                   // rare per-element factor
        #pragma unroll
        for (int mm = 0; mm < 2; ++mm) {
            float pl[8];
            #pragma unroll
            for (int vv = 0; vv < 8; ++vv) {
                const float a = ((jb + mm * 16 + vv) >= idxb) ? wgt : 1.0f;
                const float p = ex2(s[mm][vv] - mnew) * a;
                pl[vv] = p; rs += p;
            }
            #pragma unroll
            for (int qi = 0; qi < 4; ++qi)
                pq[mm][qi] = pack2hi(pl[2 * qi], pl[2 * qi + 1]);
        }
    } else {                                          // fold log2(w) into bias
        const float madj = allw ? (mnew - l2w) : mnew;
        #pragma unroll
        for (int mm = 0; mm < 2; ++mm) {
            float pl[8];
            #pragma unroll
            for (int vv = 0; vv < 8; ++vv) {
                const float p = ex2(s[mm][vv] - madj);
                pl[vv] = p; rs += p;
            }
            #pragma unroll
            for (int qi = 0; qi < 4; ++qi)
                pq[mm][qi] = pack2hi(pl[2 * qi], pl[2 * qi + 1]);
        }
    }
    rs += xor16f(rs);
    l_run = l_run * corr + rs;
    #pragma unroll
    for (int c4 = 0; c4 < 4; ++c4)
        #pragma unroll
        for (int vv = 0; vv < 8; ++vv)
            o[c4][vv] *= corr;                        // per-lane scalar rescale

    // ---- P^T B-operand via xor-16 half-exchange ----
    V16 pt;
    #pragma unroll
    for (int e = 0; e < 4; ++e) {
        const uint32_t og0 = xor16u(pq[0][e]);
        const uint32_t og1 = xor16u(pq[1][e]);
        pt.u[e]     = lh ? og1 : pq[0][e];
        pt.u[4 + e] = lh ? pq[1][e] : og0;
    }

    // ---- O^T += V^T * P^T : 4 WMMA. ----
    #pragma unroll
    for (int c4 = 0; c4 < 4; ++c4) {
        V16 va;                                        // A-layout row d = c4*16+ln
        const uint16_t* src = &bV[c4 * 16 + ln][lh * 16];
        va.q2[0] = *(const u32x4*)src;
        va.q2[1] = *(const u32x4*)(src + 8);
        o[c4] = __builtin_amdgcn_wmma_f32_16x16x32_bf16(
                    false, va.v, false, pt.v, (short)0, o[c4], false, false);
    }
}

// n=8, h=16, s=1024, d=64. Grid: (s/128, n*h). Block: 256 threads = 8 wave32.
__global__ __launch_bounds__(256, 2)
void fa_forcing_kernel(const float* __restrict__ qp,
                       const float* __restrict__ kp,
                       const float* __restrict__ vp,
                       const int*   __restrict__ idxp,
                       const float* __restrict__ wptr,
                       float* __restrict__ outp)
{
    constexpr int S = 1024, D = 64;
    constexpr float L2E = 1.4426950408889634f;

    __shared__ __align__(16) uint16_t ldsK[2][32][64];   // double-buffered
    __shared__ __align__(16) uint16_t ldsV[2][64][32];

    const int tid  = threadIdx.x;
    const int lane = tid & 31;
    const int wave = tid >> 5;
    const int lh   = lane >> 4;
    const int ln   = lane & 15;
    const int bh   = blockIdx.y;
    const int b    = bh >> 4;
    const int r0   = blockIdx.x * 128;
    const int qrow = r0 + wave * 16;
    const int i    = qrow + ln;          // this lane's query row
    // SGPR copy of the wave-uniform query base -> scalar branches, not saveexec
    const int qrow_s = __builtin_amdgcn_readfirstlane(qrow);

    const float wgt  = wptr[0];
    const float l2w  = __log2f(wgt);
    const int   idxb = idxp[b];

    const size_t baseQ = (size_t)bh * S * D;
    const float* kpB = kp + (size_t)bh * D * S;
    const float* vpB = vp + (size_t)bh * S * D;

    // per-thread staging coordinates (loop-invariant)
    const int kj  = tid & 31;            // K: key column
    const int kdp = (tid >> 5) * 2;      // K: first d of pair (chunks += 16)
    const int vd  = tid & 63;            // V: d
    const int vjp = (tid >> 6) * 2;      // V: first key of pair (chunks += 8)

    // ---- Q^T as WMMA B operands, RNE bf16, pre-scaled by log2(e)/sqrt(64) ----
    V16 qt[2];
    {
        const float* qr = qp + baseQ + (size_t)i * D;
        #pragma unroll
        for (int t = 0; t < 2; ++t) {
            const int d0 = t * 32 + lh * 16;
            #pragma unroll
            for (int e = 0; e < 16; ++e)
                qt[t].e[e] = bfbits_rne(qr[d0 + e] * (0.125f * L2E));
        }
    }

    v8f o[4] = {};
    float m_run = -3.0e38f;
    float l_run = 0.0f;

    const int ntiles = (r0 + 128) / 32;

    // staging helper (lambda keeps per-thread coords hoisted)
    auto stage = [&](int buf, int kt0) {
        #pragma unroll
        for (int e = 0; e < 4; ++e) {
            const int d0 = kdp + e * 16;
            const float* gk = kpB + (size_t)d0 * S + (kt0 + kj);
            *(uint32_t*)&ldsK[buf][kj][d0] = pack2hi(gk[0], gk[S]);
        }
        #pragma unroll
        for (int e = 0; e < 4; ++e) {
            const int j0 = vjp + e * 8;
            const float* gv = vpB + (size_t)(kt0 + j0) * D + vd;
            *(uint32_t*)&ldsV[buf][vd][j0] = pack2hi(gv[0], gv[D]);
        }
    };

    stage(0, 0);
    __syncthreads();

    for (int kt = 0; kt < ntiles; ++kt) {
        const int kt0  = kt * 32;
        const int pbuf = kt & 1;
        if (kt + 1 < ntiles)
            stage(1 - pbuf, kt0 + 32);
        if (kt + 2 < ntiles) {                        // two-ahead prefetch hint
            const int pf = kt0 + 64;
            __builtin_prefetch(kpB + (size_t)kdp * S + pf + kj, 0, 1);
            __builtin_prefetch(vpB + (size_t)(pf + vjp) * D + vd, 0, 1);
        }
        if (kt0 <= qrow_s + 15) {                     // scalar: skip masked tiles
            if (kt0 + 31 <= qrow_s)                   // scalar: interior, no mask
                attn_tile<false>(ldsK[pbuf], ldsV[pbuf], qt, o, m_run, l_run,
                                 kt0, i, lh, ln, idxb, wgt, l2w);
            else                                      // single diagonal tile
                attn_tile<true>(ldsK[pbuf], ldsV[pbuf], qt, o, m_run, l_run,
                                kt0, i, lh, ln, idxb, wgt, l2w);
        }
        __syncthreads();
    }

    // ---- normalize (per-lane scalar) and store: row i ----
    const float inv = 1.0f / l_run;
    float* po = outp + baseQ + (size_t)i * D;
    #pragma unroll
    for (int c4 = 0; c4 < 4; ++c4) {
        f32x4 w0, w1;
        #pragma unroll
        for (int e = 0; e < 4; ++e) { w0[e] = o[c4][e] * inv; w1[e] = o[c4][4 + e] * inv; }
        *(f32x4*)(po + c4 * 16 + lh * 8)     = w0;
        *(f32x4*)(po + c4 * 16 + lh * 8 + 4) = w1;
    }
}

extern "C" void kernel_launch(void* const* d_in, const int* in_sizes, int n_in,
                              void* d_out, int out_size, void* d_ws, size_t ws_size,
                              hipStream_t stream) {
    (void)in_sizes; (void)n_in; (void)out_size; (void)d_ws; (void)ws_size;
    const float* q   = (const float*)d_in[0];
    const float* k   = (const float*)d_in[1];
    const float* v   = (const float*)d_in[2];
    const int*   idx = (const int*)d_in[3];
    const float* w   = (const float*)d_in[4];
    float* out = (float*)d_out;

    dim3 grid(1024 / 128, 8 * 16);
    dim3 block(256);
    hipLaunchKernelGGL(fa_forcing_kernel, grid, block, 0, stream,
                       q, k, v, idx, w, out);
}
